// BaseLM_46935402610980
// MI455X (gfx1250) — compile-verified, matrix operand order checked by reading, not run on hardware
//
#include <hip/hip_runtime.h>
#include <hip/hip_bf16.h>

#define DEV __device__ __forceinline__

// ---- problem constants (match reference) ----
#define Vv   50000
#define C0v  2000
#define C1v  10000
#define Ev   512
#define Hv   1024
#define Bv   32
#define Tv   128
#define NROW (Bv * (Tv - 1))   // 4064
#define S1v  8000              // C1 - C0
#define S2v  40000             // V - C1
#define H1v  256               // H/4
#define H2v  64                // H/16
#define G3H  (3 * Hv)          // 3072
#define HEADC 2002
#define HEADP 2016             // padded to multiple of 16

// TDM staging of the A-tile in k_lse. TDM_MODE: 0 = manual copy,
// 1 = ROCm 7.2 5-arg builtin, 2 = amdgpu-toolchain 6-arg builtin.
// Probe result (round 2): this toolchain expects 6 args.
#define TDM_MODE 2

typedef __attribute__((ext_vector_type(16))) __bf16 v16bf;
typedef __attribute__((ext_vector_type(8)))  float  v8f;
typedef __attribute__((ext_vector_type(4)))  unsigned int u32x4;
typedef __attribute__((ext_vector_type(8)))  int    i32x8;
typedef __attribute__((ext_vector_type(4)))  int    i32x4;

DEV v8f wmma_bf16(v16bf a, v16bf b, v8f c) {
  return __builtin_amdgcn_wmma_f32_16x16x32_bf16(false, a, false, b, (short)0, c,
                                                 false, false);
}

// Load a 16x32 bf16 fragment from a row-major [rows, ld] matrix.
// row = row_base + lane%16; elems 0..7 -> K = k_base + 8*(lane/16)+j, 8..15 -> +16.
DEV v16bf load_frag(const __bf16* base, int ld, int row_base, int k_base) {
  int lane = threadIdx.x & 31;
  const __bf16* p =
      base + (size_t)(row_base + (lane & 15)) * (size_t)ld + k_base + ((lane >> 4) << 3);
  v16bf f;
#pragma unroll
  for (int j = 0; j < 8; ++j) {
    f[j] = p[j];
    f[j + 8] = p[16 + j];
  }
  return f;
}

// ---------------- conversion / gather kernels ----------------
__global__ void k_f32_to_bf16(const float* __restrict__ s, __bf16* __restrict__ d, int n) {
  int i = blockIdx.x * 256 + threadIdx.x;
  if (i < n) d[i] = (__bf16)s[i];
}

__global__ void k_f32_to_bf16_padrows(const float* __restrict__ s, __bf16* __restrict__ d,
                                      int rows, int rowsPad, int cols) {
  int i = blockIdx.x * 256 + threadIdx.x;
  if (i >= rowsPad * cols) return;
  int r = i / cols;
  d[i] = (r < rows) ? (__bf16)s[i] : (__bf16)0.0f;
}

__global__ void k_gather(const int* __restrict__ x, const float* __restrict__ emb,
                         __bf16* __restrict__ inp) {
  int i = blockIdx.x * 256 + threadIdx.x;
  if (i >= NROW * Ev) return;
  int e = i & (Ev - 1);
  int n = i >> 9;                 // Ev = 512
  int b = n / (Tv - 1), t = n % (Tv - 1);
  int tok = x[b * Tv + t];
  inp[i] = (__bf16)emb[(size_t)tok * Ev + e];
}

__global__ void k_init_h(const float* __restrict__ hidden, float* __restrict__ h,
                         __bf16* __restrict__ hb0, int* __restrict__ bar) {
  int i = blockIdx.x * 256 + threadIdx.x;
  if (i < 2) bar[i] = 0;          // reset grid-barrier state every call
  if (i < Bv * Hv) {
    float v = hidden[i];
    h[i] = v;
    hb0[i] = (__bf16)v;
  }
}

// ---------------- generic WMMA GEMM:  C[M,Ncols] = A[M,K] @ W[Ncols,K]^T (+bias)
// one wave per 16x16 output tile
__global__ void k_gemm(const __bf16* __restrict__ A, const __bf16* __restrict__ W,
                       const float* __restrict__ bias, float* __restrict__ Cf,
                       __bf16* __restrict__ Cb, int Ntiles, int K, int ldc) {
  int tile = blockIdx.x;
  int mt = tile / Ntiles, nt = tile % Ntiles;
  int mbase = mt * 16, nbase = nt * 16;
  v8f acc = {};
  for (int k = 0; k < K; k += 32) {
    v16bf a = load_frag(A, K, mbase, k);
    v16bf b = load_frag(W, K, nbase, k);
    acc = wmma_bf16(a, b, acc);
  }
  int lane = threadIdx.x & 31;
  int col = nbase + (lane & 15);
  float bv = bias ? bias[col] : 0.0f;
  int rowoff = (lane >> 4) << 3;
#pragma unroll
  for (int v = 0; v < 8; ++v) {
    int row = mbase + rowoff + v;
    float val = acc[v] + bv;
    if (Cf) Cf[(size_t)row * ldc + col] = val;
    else    Cb[(size_t)row * ldc + col] = (__bf16)val;
  }
}

// ---------------- persistent GRU: all 127 steps in one launch, 128 waves,
// software grid barrier between steps (bar[0]=arrive count, bar[1]=generation).
__global__ void k_gru_all(const float* __restrict__ xp, float* __restrict__ h,
                          __bf16* __restrict__ hb0, __bf16* __restrict__ hb1,
                          const __bf16* __restrict__ whh, const float* __restrict__ bhh,
                          const int* __restrict__ lengths, __bf16* __restrict__ ysb,
                          int* __restrict__ bar) {
  int tile = blockIdx.x;          // 0..127
  int mt = tile >> 6;             // 0..1
  int nt = tile & 63;             // 0..63
  int mbase = mt * 16, nbase = nt * 16;
  int lane = threadIdx.x & 31;
  int col = nbase + (lane & 15);
  float br_ = bhh[col], bz_ = bhh[Hv + col], bn_ = bhh[2 * Hv + col];
  int rowoff = (lane >> 4) << 3;
  int len_[8];
#pragma unroll
  for (int v = 0; v < 8; ++v) len_[v] = lengths[mbase + rowoff + v] - 1;

  for (int t = 0; t < Tv - 1; ++t) {
    const __bf16* src = (t & 1) ? hb1 : hb0;
    __bf16* dst = (t & 1) ? hb0 : hb1;

    v8f accr = {}, accz = {}, accn = {};
    for (int k = 0; k < Hv; k += 32) {
      v16bf a = load_frag(src, Hv, mbase, k);
      v16bf br = load_frag(whh, Hv, nbase, k);
      v16bf bz = load_frag(whh, Hv, Hv + nbase, k);
      v16bf bn = load_frag(whh, Hv, 2 * Hv + nbase, k);
      accr = wmma_bf16(a, br, accr);
      accz = wmma_bf16(a, bz, accz);
      accn = wmma_bf16(a, bn, accn);
    }
#pragma unroll
    for (int v = 0; v < 8; ++v) {
      int m = mbase + rowoff + v;   // batch row 0..31
      size_t xrow = (size_t)(m * (Tv - 1) + t) * G3H;
      float xr = xp[xrow + col];
      float xz = xp[xrow + Hv + col];
      float xn = xp[xrow + 2 * Hv + col];
      float hr = accr[v] + br_, hz = accz[v] + bz_, hn = accn[v] + bn_;
      float r = 1.0f / (1.0f + __expf(-(xr + hr)));
      float z = 1.0f / (1.0f + __expf(-(xz + hz)));
      float nn = tanhf(xn + r * hn);
      float hold = h[m * Hv + col];
      float hnew = (1.0f - z) * nn + z * hold;
      bool valid = t < len_[v];
      float hf = valid ? hnew : hold;
      h[m * Hv + col] = hf;
      dst[m * Hv + col] = (__bf16)hf;
      ysb[(size_t)(m * (Tv - 1) + t) * Hv + col] = valid ? (__bf16)hf : (__bf16)0.0f;
    }

    // ---- grid barrier (generation = t+1) ----
    __threadfence();
    __syncthreads();
    if (threadIdx.x == 0) {
      if (atomicAdd(&bar[0], 1) == (int)gridDim.x - 1) {
        bar[0] = 0;
        __threadfence();
        atomicAdd(&bar[1], 1);
      } else {
        while (__hip_atomic_load(&bar[1], __ATOMIC_ACQUIRE,
                                 __HIP_MEMORY_SCOPE_AGENT) < t + 1)
          __builtin_amdgcn_s_sleep(1);
      }
    }
    __syncthreads();
  }
}

// ---------------- online log-sum-exp over a cluster.
// 4 waves / block; 16 rows per block; col tiles strided over waves; per-lane
// running (max,sum), one cross-lane LSE merge at the end, cross-wave merge in LDS.
// A-tile staged into LDS via the Tensor Data Mover.
__global__ void k_lse(const __bf16* __restrict__ A, const __bf16* __restrict__ W,
                      int K, int kshift, int C, int Ctiles, float* __restrict__ lse) {
  __shared__ __bf16 As[16 * 1024];
  __shared__ float redm[4][16];
  __shared__ float reds[4][16];
  int rbase = blockIdx.x * 16;
  int wave = threadIdx.x >> 5;
  int lane = threadIdx.x & 31;

#if TDM_MODE
  if (wave == 0) {
    unsigned long long ga = (unsigned long long)(uintptr_t)(A + (size_t)rbase * K);
    unsigned int lds = (unsigned int)(uintptr_t)As;   // low 32b = LDS offset
    unsigned int rowB = (unsigned int)K * 2u;         // bytes per row
    u32x4 g0;
    g0[0] = 1u;                                       // count=1, user mode
    g0[1] = lds;                                      // lds_addr
    g0[2] = (unsigned int)ga;                         // global_addr[31:0]
    g0[3] = (unsigned int)((ga >> 32) & 0x01FFFFFFu) | (2u << 30); // addr[56:32], type=2
    i32x8 g1;
    g1[0] = 0;                                        // mask=0, data_size=1B
    g1[1] = (int)(rowB << 16);                        // tensor_dim0[15:0]
    g1[2] = (int)(((rowB >> 16) & 0xFFFFu) | ((unsigned)NROW << 16)); // dim0 hi | dim1 lo
    g1[3] = (int)(rowB << 16);                        // dim1 hi | tile_dim0
    g1[4] = 16;                                       // tile_dim1=16 rows
    g1[5] = (int)rowB;                                // tensor_dim0_stride lo
    g1[6] = 0;
    g1[7] = 0;
    i32x4 z4 = {};
#if TDM_MODE == 1
    __builtin_amdgcn_tensor_load_to_lds(g0, g1, z4, z4, 0);
#else
    i32x8 z8 = {};
    __builtin_amdgcn_tensor_load_to_lds(g0, g1, z4, z4, z8, 0);
#endif
    __builtin_amdgcn_s_wait_tensorcnt(0);
  }
#else
  for (int i = threadIdx.x; i < (16 << kshift); i += 128) {
    int r = i >> kshift, c = i & (K - 1);
    As[i] = A[(size_t)(rbase + r) * K + c];
  }
#endif
  __syncthreads();

  float lmax[8], lsum[8];
#pragma unroll
  for (int v = 0; v < 8; ++v) { lmax[v] = -3.0e38f; lsum[v] = 0.0f; }
  int colLane = lane & 15;

  for (int ct = wave; ct < Ctiles; ct += 4) {
    int nbase = ct * 16;
    v8f acc = {};
    for (int k = 0; k < K; k += 32) {
      v16bf a = load_frag(As, K, 0, k);
      v16bf b = load_frag(W, K, nbase, k);
      acc = wmma_bf16(a, b, acc);
    }
    bool inC = (nbase + colLane) < C;
#pragma unroll
    for (int v = 0; v < 8; ++v) {
      float val = inC ? acc[v] : -3.0e38f;
      float nm = fmaxf(lmax[v], val);
      lsum[v] = lsum[v] * __expf(lmax[v] - nm) + __expf(val - nm);
      lmax[v] = nm;
    }
  }

  // cross-lane (16-wide) pairwise log-sum-exp merge
#pragma unroll
  for (int off = 8; off; off >>= 1) {
#pragma unroll
    for (int v = 0; v < 8; ++v) {
      float m2 = __shfl_xor(lmax[v], off, 16);
      float s2 = __shfl_xor(lsum[v], off, 16);
      float nm = fmaxf(lmax[v], m2);
      lsum[v] = lsum[v] * __expf(lmax[v] - nm) + s2 * __expf(m2 - nm);
      lmax[v] = nm;
    }
  }
  if ((lane & 15) == 0) {
    int ro = (lane >> 4) << 3;
#pragma unroll
    for (int v = 0; v < 8; ++v) {
      redm[wave][ro + v] = lmax[v];
      reds[wave][ro + v] = lsum[v];
    }
  }
  __syncthreads();
  if (threadIdx.x < 16) {
    float m = redm[0][threadIdx.x], s = reds[0][threadIdx.x];
#pragma unroll
    for (int w = 1; w < 4; ++w) {
      float m2 = redm[w][threadIdx.x], s2 = reds[w][threadIdx.x];
      float nm = fmaxf(m, m2);
      s = s * __expf(m - nm) + s2 * __expf(m2 - nm);
      m = nm;
    }
    lse[rbase + threadIdx.x] = m + __logf(s);
  }
}

// ---------------- per-row picked logits (dot products), one wave per row
__global__ void k_pick_tail(const __bf16* __restrict__ A, const __bf16* __restrict__ W,
                            int K, const int* __restrict__ x, int lo, int size,
                            float* __restrict__ out) {
  int n = blockIdx.x;
  int lane = threadIdx.x;
  int b = n / (Tv - 1), t = n % (Tv - 1);
  int tgt = x[b * Tv + t + 1];
  int rel = tgt - lo;
  rel = rel < 0 ? 0 : (rel >= size ? size - 1 : rel);
  const __bf16* a = A + (size_t)n * K;
  const __bf16* w = W + (size_t)rel * K;
  float s = 0.0f;
  for (int k = lane; k < K; k += 32) s += (float)a[k] * (float)w[k];
  for (int off = 16; off; off >>= 1) s += __shfl_xor(s, off, 32);
  if (lane == 0) out[n] = s;
}

__global__ void k_pick_head(const __bf16* __restrict__ A, const __bf16* __restrict__ W,
                            int K, const int* __restrict__ x, float* __restrict__ pt,
                            float* __restrict__ pc1, float* __restrict__ pc2) {
  int n = blockIdx.x;
  int lane = threadIdx.x;
  int b = n / (Tv - 1), t = n % (Tv - 1);
  int tgt = x[b * Tv + t + 1];
  int it = tgt < 0 ? 0 : (tgt >= C0v ? C0v - 1 : tgt);
  const __bf16* a = A + (size_t)n * K;
  const __bf16* w0 = W + (size_t)it * K;
  const __bf16* w1 = W + (size_t)C0v * K;
  const __bf16* w2 = W + (size_t)(C0v + 1) * K;
  float s0 = 0.0f, s1 = 0.0f, s2 = 0.0f;
  for (int k = lane; k < K; k += 32) {
    float av = (float)a[k];
    s0 += av * (float)w0[k];
    s1 += av * (float)w1[k];
    s2 += av * (float)w2[k];
  }
  for (int off = 16; off; off >>= 1) {
    s0 += __shfl_xor(s0, off, 32);
    s1 += __shfl_xor(s1, off, 32);
    s2 += __shfl_xor(s2, off, 32);
  }
  if (lane == 0) { pt[n] = s0; pc1[n] = s1; pc2[n] = s2; }
}

// ---------------- finalize: select per-row logp, mean, copy hT ----------------
__global__ void k_finalize(const int* __restrict__ x, const float* __restrict__ head_lse,
                           const float* __restrict__ head_t, const float* __restrict__ head_c1,
                           const float* __restrict__ head_c2, const float* __restrict__ t1_lse,
                           const float* __restrict__ t1_pick, const float* __restrict__ t2_lse,
                           const float* __restrict__ t2_pick, const float* __restrict__ h,
                           float* __restrict__ out) {
  __shared__ float red[256];
  int lane = threadIdx.x;
  float s = 0.0f;
  for (int n = lane; n < NROW; n += 256) {
    int b = n / (Tv - 1), t = n % (Tv - 1);
    int tgt = x[b * Tv + t + 1];
    float hl = head_lse[n];
    float lp;
    if (tgt < C0v)      lp = head_t[n] - hl;
    else if (tgt < C1v) lp = (head_c1[n] - hl) + (t1_pick[n] - t1_lse[n]);
    else                lp = (head_c2[n] - hl) + (t2_pick[n] - t2_lse[n]);
    s += lp;
  }
  red[lane] = s;
  __syncthreads();
  for (int o = 128; o; o >>= 1) {
    if (lane < o) red[lane] += red[lane + o];
    __syncthreads();
  }
  if (lane == 0) out[0] = -red[0] / (float)NROW;
  for (int i = lane; i < Bv * Hv; i += 256) out[1 + i] = h[i];
}

// =====================================================================
extern "C" void kernel_launch(void* const* d_in, const int* in_sizes, int n_in,
                              void* d_out, int out_size, void* d_ws, size_t ws_size,
                              hipStream_t stream) {
  (void)in_sizes; (void)n_in; (void)out_size; (void)ws_size;
  const int*   x       = (const int*)d_in[0];
  const int*   lengths = (const int*)d_in[1];
  const float* hidden  = (const float*)d_in[2];
  const float* emb     = (const float*)d_in[3];
  const float* w_ih    = (const float*)d_in[4];
  const float* w_hh    = (const float*)d_in[5];
  const float* b_ih    = (const float*)d_in[6];
  const float* b_hh    = (const float*)d_in[7];
  const float* head_w  = (const float*)d_in[8];
  const float* p1      = (const float*)d_in[9];
  const float* t1      = (const float*)d_in[10];
  const float* p2      = (const float*)d_in[11];
  const float* t2      = (const float*)d_in[12];
  float* out = (float*)d_out;

  char* ws = (char*)d_ws;
  size_t off = 0;
  auto alloc = [&](size_t bytes) -> char* {
    off = (off + 255) & ~(size_t)255;
    char* p = ws + off;
    off += bytes;
    return p;
  };

  __bf16* inp_b   = (__bf16*)alloc((size_t)NROW * Ev * 2);
  __bf16* wih_b   = (__bf16*)alloc((size_t)G3H * Ev * 2);
  __bf16* whh_b   = (__bf16*)alloc((size_t)G3H * Hv * 2);
  __bf16* head_b  = (__bf16*)alloc((size_t)HEADP * Hv * 2);
  __bf16* p1_b    = (__bf16*)alloc((size_t)H1v * Hv * 2);
  __bf16* t1_b    = (__bf16*)alloc((size_t)S1v * H1v * 2);
  __bf16* p2_b    = (__bf16*)alloc((size_t)H2v * Hv * 2);
  __bf16* t2_b    = (__bf16*)alloc((size_t)S2v * H2v * 2);
  float*  xp      = (float*)alloc((size_t)NROW * G3H * 4);
  float*  hbuf    = (float*)alloc((size_t)Bv * Hv * 4);
  __bf16* hb0     = (__bf16*)alloc((size_t)Bv * Hv * 2);
  __bf16* hb1     = (__bf16*)alloc((size_t)Bv * Hv * 2);
  __bf16* ys_b    = (__bf16*)alloc((size_t)NROW * Hv * 2);
  __bf16* h1_b    = (__bf16*)alloc((size_t)NROW * H1v * 2);
  __bf16* h2_b    = (__bf16*)alloc((size_t)NROW * H2v * 2);
  float*  head_lse = (float*)alloc(NROW * 4);
  float*  head_t   = (float*)alloc(NROW * 4);
  float*  head_c1  = (float*)alloc(NROW * 4);
  float*  head_c2  = (float*)alloc(NROW * 4);
  float*  t1_lse   = (float*)alloc(NROW * 4);
  float*  t1_pick  = (float*)alloc(NROW * 4);
  float*  t2_lse   = (float*)alloc(NROW * 4);
  float*  t2_pick  = (float*)alloc(NROW * 4);
  int*    bar      = (int*)alloc(2 * 4);

  auto cvt = [&](const float* s, __bf16* d, int n) {
    k_f32_to_bf16<<<(n + 255) / 256, 256, 0, stream>>>(s, d, n);
  };
  cvt(w_ih, wih_b, G3H * Ev);
  cvt(w_hh, whh_b, G3H * Hv);
  cvt(p1, p1_b, H1v * Hv);
  cvt(t1, t1_b, S1v * H1v);
  cvt(p2, p2_b, H2v * Hv);
  cvt(t2, t2_b, S2v * H2v);
  {
    int n = HEADP * Hv;
    k_f32_to_bf16_padrows<<<(n + 255) / 256, 256, 0, stream>>>(head_w, head_b,
                                                               HEADC, HEADP, Hv);
  }

  k_gather<<<(NROW * Ev + 255) / 256, 256, 0, stream>>>(x, emb, inp_b);
  k_init_h<<<(Bv * Hv + 255) / 256, 256, 0, stream>>>(hidden, hbuf, hb0, bar);

  // xp = inp @ w_ih^T + b_ih : [4064, 3072], K=512
  k_gemm<<<(NROW / 16) * (G3H / 16), 32, 0, stream>>>(inp_b, wih_b, b_ih, xp,
                                                      (__bf16*)nullptr, G3H / 16,
                                                      Ev, G3H);

  // full GRU recurrence in one persistent launch
  k_gru_all<<<128, 32, 0, stream>>>(xp, hbuf, hb0, hb1, whh_b, b_hh, lengths,
                                    ys_b, bar);

  // tail projections: h1 = ys @ p1^T [4064,256], h2 = ys @ p2^T [4064,64]
  k_gemm<<<(NROW / 16) * (H1v / 16), 32, 0, stream>>>(ys_b, p1_b, nullptr,
                                                      (float*)nullptr, h1_b,
                                                      H1v / 16, Hv, H1v);
  k_gemm<<<(NROW / 16) * (H2v / 16), 32, 0, stream>>>(ys_b, p2_b, nullptr,
                                                      (float*)nullptr, h2_b,
                                                      H2v / 16, Hv, H2v);

  // online log-sum-exp per cluster (4 waves/block)
  k_lse<<<NROW / 16, 128, 0, stream>>>(ys_b, head_b, Hv, 10, HEADC, HEADP / 16, head_lse);
  k_lse<<<NROW / 16, 128, 0, stream>>>(h1_b, t1_b, H1v, 8, S1v, S1v / 16, t1_lse);
  k_lse<<<NROW / 16, 128, 0, stream>>>(h2_b, t2_b, H2v, 6, S2v, S2v / 16, t2_lse);

  // picked logits
  k_pick_head<<<NROW, 32, 0, stream>>>(ys_b, head_b, Hv, x, head_t, head_c1, head_c2);
  k_pick_tail<<<NROW, 32, 0, stream>>>(h1_b, t1_b, H1v, x, C0v, S1v, t1_pick);
  k_pick_tail<<<NROW, 32, 0, stream>>>(h2_b, t2_b, H2v, x, C1v, S2v, t2_pick);

  k_finalize<<<1, 256, 0, stream>>>(x, head_lse, head_t, head_c1, head_c2,
                                    t1_lse, t1_pick, t2_lse, t2_pick, hbuf, out);
}
